// RGCN_18262200943184
// MI455X (gfx1250) — matrix operand order repaired and checked by reading, not compiled, not run on hardware
//
#include <hip/hip_runtime.h>
#include <hip/hip_bf16.h>

typedef __attribute__((ext_vector_type(16))) _Float16 v16h;
typedef __attribute__((ext_vector_type(8)))  float    v8f;

#define MB_  32
#define E_   4
#define NN_  512
#define INH_ 64
#define HID_ 128
#define CAT_ 640   // HID + HID*E
#define TILE_HALFS 512  // 32 lanes * 16 halves per WMMA operand tile

// ============================================================================
// Packed operand layouts (CDNA5 wave32 WMMA VGPR striping, ISA 7.12.2,
// serialized lane-major so a wave loads one operand tile as 1KB contiguous):
//   A tile (16x32 f16): lanes 0-15 row=lane, halves[0..7]=K0..7,[8..15]=K16..23
//                       lanes 16-31 row=lane-16, halves=K8..15,K24..31
//   B tile (32x16 f16): lanes 0-15 col=lane, halves[0..7]=K0..7,[8..15]=K16..23
//                       lanes 16-31 col=lane-16, halves=K8..15,K24..31
// Tile order: A -> [mt][kt], B -> [nt][kt]  (kt innermost = streaming k-loop)
// ============================================================================

__device__ __forceinline__ size_t apack_off(int row, int col, int KT) {
  int mt = row >> 4, r = row & 15;
  int kt = col >> 5, kin = col & 31;
  int lane = r + (((kin >> 3) & 1) << 4);
  int i = (kin & 7) + ((kin >> 4) << 3);
  return ((size_t)(mt * KT + kt) * 32 + lane) * 16 + i;
}

__device__ __forceinline__ size_t bpack_off(int k, int n, int KT) {
  int kt = k >> 5, kin = k & 31;
  int nt = n >> 4, c = n & 15;
  int lane = c + (((kin >> 3) & 1) << 4);
  int i = (kin & 7) + ((kin >> 4) << 3);
  return ((size_t)(nt * KT + kt) * 32 + lane) * 16 + i;
}

__device__ __forceinline__ v16h ld_tile(const _Float16* __restrict__ p) {
  return *(const v16h*)p;   // 32B aligned -> 2x global_load_b128
}

#define WMMA_F16(A_, B_, C_) \
  __builtin_amdgcn_wmma_f32_16x16x32_f16(false, (A_), false, (B_), (short)0, (C_), false, false)

// ---------------- generic packed f16 WMMA GEMM, 2x4 wave tiling ---------------
// Wave computes a 32x64 output patch. C(f32, row-major MxN) = A @ B + bias.
__global__ __launch_bounds__(256)
void gemm_f16_wmma(const _Float16* __restrict__ A, const _Float16* __restrict__ B,
                   const float* __restrict__ bias, float* __restrict__ C,
                   int M, int N, int K) {
  int lane = threadIdx.x & 31;
  int wave = threadIdx.x >> 5;
  int KT = K >> 5;
  int n4 = N >> 6;                               // 64-wide column supertiles
  int tile = blockIdx.x * 8 + wave;
  int mt2 = tile / n4, ntq = tile - mt2 * n4;
  if (mt2 >= (M >> 5)) return;                   // uniform per wave
  const _Float16* A0 = A + (size_t)(2 * mt2) * KT * TILE_HALFS;
  const _Float16* A1 = A0 + (size_t)KT * TILE_HALFS;
  const _Float16* Bb[4];
#pragma unroll
  for (int j = 0; j < 4; ++j)
    Bb[j] = B + (size_t)(ntq * 4 + j) * KT * TILE_HALFS;
  size_t lo = (size_t)lane * 16;
  v8f acc0[4] = {}, acc1[4] = {};
  for (int kt = 0; kt < KT; ++kt) {
    size_t o = (size_t)kt * TILE_HALFS + lo;
    v16h a0 = ld_tile(A0 + o);
    v16h a1 = ld_tile(A1 + o);
#pragma unroll
    for (int j = 0; j < 4; ++j) {
      v16h bj = ld_tile(Bb[j] + o);
      acc0[j] = WMMA_F16(a0, bj, acc0[j]);
      acc1[j] = WMMA_F16(a1, bj, acc1[j]);
    }
  }
  int c  = lane & 15;
  int mo = (lane >> 4) << 3;                     // D: rows 0-7 / 8-15 per half-wave
  int mr = (mt2 << 5) + mo;
  int nc = (ntq << 6);
  float* Cp = C + (size_t)mr * N + nc + c;
#pragma unroll
  for (int j = 0; j < 4; ++j) {
    float bv = bias ? bias[nc + 16 * j + c] : 0.f;
#pragma unroll
    for (int r = 0; r < 8; ++r) {
      Cp[(size_t)r * N + 16 * j]        = acc0[j][r] + bv;
      Cp[(size_t)(r + 16) * N + 16 * j] = acc1[j][r] + bv;
    }
  }
}

// -------- relational aggregation: hconv[b] = hs[b] + sum_e na[b,e] @ m[b,e] ----
// na packed A per (b,e) 512x512 (KT=16); mf packed B per (b,e) 512x128 (KT=16).
// 2x4 wave tiling (32x64 patch); 32 wave-tiles/batch -> grid (4, MB).
__global__ __launch_bounds__(256)
void rgcn_agg_wmma(const _Float16* __restrict__ na, const _Float16* __restrict__ mf,
                   const float* __restrict__ hsm, float* __restrict__ hconv) {
  const int KT = NN_ >> 5;                       // 16
  int lane = threadIdx.x & 31;
  int wave = threadIdx.x >> 5;
  int b = blockIdx.y;
  int tile = blockIdx.x * 8 + wave;              // 0..31
  int mt2 = tile >> 1, ntq = tile & 1;           // 16 row x 2 col supertiles
  size_t lo = (size_t)lane * 16;
  v8f acc0[4] = {}, acc1[4] = {};
  for (int e = 0; e < E_; ++e) {
    const _Float16* A0 = na + (size_t)(b * E_ + e) * NN_ * NN_
                            + (size_t)(2 * mt2) * KT * TILE_HALFS;
    const _Float16* A1 = A0 + (size_t)KT * TILE_HALFS;
    const _Float16* Bbase = mf + (size_t)(b * E_ + e) * NN_ * HID_;
    const _Float16* Bb[4];
#pragma unroll
    for (int j = 0; j < 4; ++j)
      Bb[j] = Bbase + (size_t)(ntq * 4 + j) * KT * TILE_HALFS;
    for (int kt = 0; kt < KT; ++kt) {
      size_t o = (size_t)kt * TILE_HALFS + lo;
      __builtin_prefetch(A0 + o + TILE_HALFS, 0, 0);   // global_prefetch_b8
      __builtin_prefetch(A1 + o + TILE_HALFS, 0, 0);
      v16h a0 = ld_tile(A0 + o);
      v16h a1 = ld_tile(A1 + o);
#pragma unroll
      for (int j = 0; j < 4; ++j) {
        v16h bj = ld_tile(Bb[j] + o);
        acc0[j] = WMMA_F16(a0, bj, acc0[j]);
        acc1[j] = WMMA_F16(a1, bj, acc1[j]);
      }
    }
  }
  int c  = lane & 15;
  int mo = (lane >> 4) << 3;
  size_t rbase = (size_t)b * NN_ + (mt2 << 5) + mo;
  int nc = (ntq << 6);
#pragma unroll
  for (int j = 0; j < 4; ++j) {
    int col = nc + 16 * j + c;
#pragma unroll
    for (int r = 0; r < 8; ++r) {
      size_t r0 = rbase + r, r1 = rbase + 16 + r;
      hconv[r0 * HID_ + col] = hsm[r0 * CAT_ + col] + acc0[j][r];
      hconv[r1 * HID_ + col] = hsm[r1 * CAT_ + col] + acc1[j][r];
    }
  }
}

// ---------------- degree + inverse (column degree over edge types & rows) ------
__global__ void deg_inv_kernel(const float* __restrict__ adj, float* __restrict__ deginv) {
  int b = blockIdx.y;
  int m = blockIdx.x * blockDim.x + threadIdx.x;
  float s = 0.f, diag = 0.f;
  for (int e = 0; e < E_; ++e) {
    const float* base = adj + (size_t)(b * E_ + e) * NN_ * NN_;
    for (int n = 0; n < NN_; ++n) s += base[(size_t)n * NN_ + m];
    diag += base[(size_t)m * NN_ + m];
  }
  float d = s - diag;
  deginv[b * NN_ + m] = d > 0.f ? 1.f / d : 0.f;
}

// -------- normalized adjacency -> fp16, written directly in packed A layout ----
__global__ void build_na_kernel(const float* __restrict__ adj, const float* __restrict__ deginv,
                                _Float16* __restrict__ na) {
  size_t idx = (size_t)blockIdx.x * blockDim.x + threadIdx.x;   // (b,e,n,m)
  int m = (int)(idx & (NN_ - 1));
  int n = (int)((idx >> 9) & (NN_ - 1));
  int b = (int)(idx >> 20);
  size_t be = idx >> 18;                                        // b*E + e
  float v = adj[idx];
  float sc = deginv[b * NN_ + m];
  na[be * NN_ * NN_ + apack_off(n, m, NN_ >> 5)] =
      (_Float16)((n == m) ? 0.f : v * sc);
}

// -------- repack edge output: m[b,e,k,o] = hsm[b,k, 128+o*E+e], packed B --------
__global__ void build_m_kernel(const float* __restrict__ hsm, _Float16* __restrict__ mf) {
  size_t idx = (size_t)blockIdx.x * blockDim.x + threadIdx.x;   // (b,e,k,o)
  int o = (int)(idx & (HID_ - 1));
  int k = (int)((idx >> 7) & (NN_ - 1));
  int e = (int)((idx >> 16) & 3);
  int b = (int)(idx >> 18);
  float v = hsm[(size_t)(b * NN_ + k) * CAT_ + HID_ + (o << 2) + e];
  mf[(size_t)(b * E_ + e) * NN_ * HID_ + bpack_off(k, o, NN_ >> 5)] = (_Float16)v;
}

// -------- concat [w_self | w_edge] -> packed B (K x 640) + bias concat ----------
__global__ void build_wcat_kernel(const float* __restrict__ wsf, const float* __restrict__ bsf,
                                  const float* __restrict__ wef, const float* __restrict__ bef,
                                  _Float16* __restrict__ wcat, float* __restrict__ bcat, int K) {
  int idx = blockIdx.x * blockDim.x + threadIdx.x;
  if (idx >= K * CAT_) return;
  int j = idx % CAT_, k = idx / CAT_;
  float v = (j < HID_) ? wsf[k * HID_ + j] : wef[k * (HID_ * E_) + (j - HID_)];
  wcat[bpack_off(k, j, K >> 5)] = (_Float16)v;
  if (k == 0) bcat[j] = (j < HID_) ? bsf[j] : bef[j - HID_];
}

// -------- row-major f32 (R x K) -> packed A fp16 --------------------------------
__global__ void cvt_apack_kernel(const float* __restrict__ in, _Float16* __restrict__ out,
                                 int R, int K) {
  size_t idx = (size_t)blockIdx.x * blockDim.x + threadIdx.x;
  if (idx >= (size_t)R * K) return;
  int row = (int)(idx / K), col = (int)(idx - (size_t)row * K);
  out[apack_off(row, col, K >> 5)] = (_Float16)in[idx];
}

// -------- row-major f32 (K x N) -> packed B fp16 --------------------------------
__global__ void cvt_bpack_kernel(const float* __restrict__ in, _Float16* __restrict__ out,
                                 int K, int N) {
  size_t idx = (size_t)blockIdx.x * blockDim.x + threadIdx.x;
  if (idx >= (size_t)K * N) return;
  int k = (int)(idx / N), n = (int)(idx - (size_t)k * N);
  out[bpack_off(k, n, K >> 5)] = (_Float16)in[idx];
}

// -------- per-node BatchNorm over (batch, channel) + ReLU -> packed A fp16 ------
__global__ __launch_bounds__(HID_)
void bn_relu_kernel(const float* __restrict__ h, const float* __restrict__ g,
                    const float* __restrict__ bb, _Float16* __restrict__ out) {
  __shared__ float rs[HID_], rs2[HID_];
  int n = blockIdx.x, d = threadIdx.x;
  float s = 0.f, s2 = 0.f;
  for (int b = 0; b < MB_; ++b) {
    float v = h[(size_t)(b * NN_ + n) * HID_ + d];
    s += v; s2 += v * v;
  }
  rs[d] = s; rs2[d] = s2;
  __syncthreads();
  for (int off = HID_ / 2; off > 0; off >>= 1) {
    if (d < off) { rs[d] += rs[d + off]; rs2[d] += rs2[d + off]; }
    __syncthreads();
  }
  const float inv_cnt = 1.f / (MB_ * HID_);
  float mean = rs[0] * inv_cnt;
  float var  = rs2[0] * inv_cnt - mean * mean;
  float inv  = rsqrtf(var + 1e-5f);
  float gg = g[n], bv = bb[n];
  for (int b = 0; b < MB_; ++b) {
    size_t off2 = (size_t)(b * NN_ + n) * HID_ + d;
    float y = (h[off2] - mean) * inv * gg + bv;
    out[apack_off(b * NN_ + n, d, HID_ >> 5)] = (_Float16)fmaxf(y, 0.f);
  }
}

extern "C" void kernel_launch(void* const* d_in, const int* in_sizes, int n_in,
                              void* d_out, int out_size, void* d_ws, size_t ws_size,
                              hipStream_t stream) {
  (void)in_sizes; (void)n_in; (void)out_size; (void)ws_size;
  const float* x   = (const float*)d_in[0];
  const float* adj = (const float*)d_in[1];
  const float* wsL[3] = {(const float*)d_in[2],  (const float*)d_in[8],  (const float*)d_in[14]};
  const float* bsL[3] = {(const float*)d_in[3],  (const float*)d_in[9],  (const float*)d_in[15]};
  const float* weL[3] = {(const float*)d_in[4],  (const float*)d_in[10], (const float*)d_in[16]};
  const float* beL[3] = {(const float*)d_in[5],  (const float*)d_in[11], (const float*)d_in[17]};
  const float* bgL[3] = {(const float*)d_in[6],  (const float*)d_in[12], (const float*)d_in[18]};
  const float* bbL[3] = {(const float*)d_in[7],  (const float*)d_in[13], (const float*)d_in[19]};
  const float* w1   = (const float*)d_in[20];
  const float* b1   = (const float*)d_in[21];
  const float* bnfg = (const float*)d_in[22];
  const float* bnfb = (const float*)d_in[23];
  const float* w2   = (const float*)d_in[24];
  const float* b2   = (const float*)d_in[25];
  float* out = (float*)d_out;

  char* ws = (char*)d_ws;
  auto alloc = [&](size_t bytes) -> char* {
    char* p = ws; ws += (bytes + 255) & ~(size_t)255; return p;
  };
  _Float16* na16  = (_Float16*)alloc((size_t)MB_ * E_ * NN_ * NN_ * 2);  // 67 MB, L2-resident
  _Float16* hf16  = (_Float16*)alloc((size_t)MB_ * NN_ * HID_ * 2);
  float*    hsm   = (float*)   alloc((size_t)MB_ * NN_ * CAT_ * 4);
  _Float16* mf16  = (_Float16*)alloc((size_t)MB_ * E_ * NN_ * HID_ * 2);
  float*    hconv = (float*)   alloc((size_t)MB_ * NN_ * HID_ * 4);
  float*    deginv= (float*)   alloc((size_t)MB_ * NN_ * 4);
  _Float16* wcat  = (_Float16*)alloc((size_t)HID_ * CAT_ * 2);
  float*    bcat  = (float*)   alloc((size_t)CAT_ * 4);
  _Float16* w1h   = (_Float16*)alloc((size_t)HID_ * HID_ * 2);
  _Float16* w2h   = (_Float16*)alloc((size_t)HID_ * HID_ * 2);

  const int M = MB_ * NN_;   // 16384 rows

  { // x -> packed A fp16 (layer-0 GEMM input, K=64)
    int n = MB_ * NN_ * INH_;
    cvt_apack_kernel<<<(n + 255) / 256, 256, 0, stream>>>(x, hf16, M, INH_);
  }
  deg_inv_kernel<<<dim3(NN_ / 256, MB_), 256, 0, stream>>>(adj, deginv);
  {
    size_t tot = (size_t)MB_ * E_ * NN_ * NN_;
    build_na_kernel<<<(unsigned)(tot / 256), 256, 0, stream>>>(adj, deginv, na16);
  }

  for (int l = 0; l < 3; ++l) {
    int K = (l == 0) ? INH_ : HID_;
    build_wcat_kernel<<<(K * CAT_ + 255) / 256, 256, 0, stream>>>(
        wsL[l], bsL[l], weL[l], beL[l], wcat, bcat, K);
    // fused self+edge projection: (16384 x K) @ (K x 640)
    gemm_f16_wmma<<<(M / 32) * (CAT_ / 64) / 8, 256, 0, stream>>>(
        hf16, wcat, bcat, hsm, M, CAT_, K);
    {
      size_t tot = (size_t)MB_ * E_ * NN_ * HID_;
      build_m_kernel<<<(unsigned)(tot / 256), 256, 0, stream>>>(hsm, mf16);
    }
    rgcn_agg_wmma<<<dim3(4, MB_), 256, 0, stream>>>(na16, mf16, hsm, hconv);
    bn_relu_kernel<<<NN_, HID_, 0, stream>>>(hconv, bgL[l], bbL[l], hf16);
  }

  cvt_bpack_kernel<<<(HID_ * HID_ + 255) / 256, 256, 0, stream>>>(w1, w1h, HID_, HID_);
  cvt_bpack_kernel<<<(HID_ * HID_ + 255) / 256, 256, 0, stream>>>(w2, w2h, HID_, HID_);
  gemm_f16_wmma<<<(M / 32) * (HID_ / 64) / 8, 256, 0, stream>>>(hf16, w1h, b1, hconv, M, HID_, HID_);
  bn_relu_kernel<<<NN_, HID_, 0, stream>>>(hconv, bnfg, bnfb, hf16);
  gemm_f16_wmma<<<(M / 32) * (HID_ / 64) / 8, 256, 0, stream>>>(hf16, w2h, b2, out, M, HID_, HID_);
}
